// Convolution_50087908606124
// MI455X (gfx1250) — matrix operand order repaired and checked by reading, not compiled, not run on hardware
//
#include <hip/hip_runtime.h>

typedef __attribute__((ext_vector_type(16))) _Float16 v16h;
typedef __attribute__((ext_vector_type(8)))  _Float16 v8h;
typedef __attribute__((ext_vector_type(8)))  float    v8f;
typedef __attribute__((ext_vector_type(4)))  float    v4f;
typedef __attribute__((ext_vector_type(4)))  unsigned int uint32x4;
typedef __attribute__((ext_vector_type(4)))  int      int32x4;
typedef __attribute__((ext_vector_type(8)))  int      int32x8;

#define N_NODES     10000
#define N_EDGES     160000
#define FC_HIDDEN   256
#define TM          256          // edges per workgroup
#define NWAVES      16           // 1 M-tile (16 edges) per wave
#define THREADS     (NWAVES * 32)
#define HSTRIDE     264          // fp16 per h row (256 + 8 pad): conflict-free A loads

#define FRAG_HALFS  512          // one B fragment = 32 lanes * 16 halfs = 1KB
#define CHUNK_TILES 4            // cu-tiles per staged chunk
#define CHUNK_FRAGS (CHUNK_TILES * 8)                 // 32 fragments
#define CHUNK_HALFS (CHUNK_FRAGS * FRAG_HALFS)        // 16384 halfs = 32KB
#define NCHUNKS     16           // 64 cu-tiles total

// dynamic LDS layout (bytes)
#define H_OFF      0
#define H_BYTES    (TM * HSTRIDE * 2)                 // 135168
#define COEF_OFF   (H_OFF + H_BYTES)
#define COEF_BYTES (6 * 16 * TM * 4)                  // 98304
#define SHY_OFF    (COEF_OFF + COEF_BYTES)
#define SHY_BYTES  (TM * 3 * 4)
#define DST_OFF    (SHY_OFF + SHY_BYTES)
#define DST_BYTES  (TM * 4)
#define W1_OFF     (DST_OFF + DST_BYTES)
#define W1_BYTES   (3 * FC_HIDDEN * 4)
#define B_OFF      (W1_OFF + W1_BYTES)
#define B_BYTES    (2 * CHUNK_HALFS * 2)              // 64KB double buffer
#define SMEM_TOTAL (B_OFF + B_BYTES)                  // 306176 < 320KB

#if __has_builtin(__builtin_amdgcn_tensor_load_to_lds)
#define HAS_TDM 1
#else
#define HAS_TDM 0
#endif

// ---------------------------------------------------------------------------
// Kernel 1: one-time conversion of fc_w2 (256x1024 f32) into fp16 WMMA
// B-fragments: per fragment (1KB), [32 lanes x 16B lo-halfs][32 lanes x 16B
// hi-halfs] -> two 2-pass bank-conflict-free ds_load_b128 on consumption.
// ---------------------------------------------------------------------------
__global__ void swizzle_w2_kernel(const float* __restrict__ w2,
                                  _Float16* __restrict__ bswz) {
  int tid  = blockIdx.x * blockDim.x + threadIdx.x;  // 0 .. 16383
  int frag = tid >> 5;                               // 0..511
  int lane = tid & 31;
  int cu   = frag >> 3;
  int kt   = frag & 7;
  int n    = cu * 16 + (lane & 15);                  // output column
  int k0   = kt * 32 + (lane >> 4) * 16;             // K base for this lane
  v16h v;
#pragma unroll
  for (int j = 0; j < 16; ++j)
    v[j] = (_Float16)w2[(size_t)(k0 + j) * 1024 + n];
  v8h lo = __builtin_shufflevector(v, v, 0, 1, 2, 3, 4, 5, 6, 7);
  v8h hi = __builtin_shufflevector(v, v, 8, 9, 10, 11, 12, 13, 14, 15);
  _Float16* base = bswz + (size_t)frag * FRAG_HALFS;
  *(v8h*)(base + lane * 8)       = lo;
  *(v8h*)(base + 256 + lane * 8) = hi;
}

// ---------------------------------------------------------------------------
// TDM: DMA one 32KB chunk of swizzled B into LDS (D# per ISA 8.3/8.4).
// ---------------------------------------------------------------------------
#define TDM_ELEMS 4096u   // 32KB / 8B
__device__ __forceinline__ void tdm_load_chunk(const _Float16* gsrc,
                                               unsigned lds_byte_off) {
#if HAS_TDM
  unsigned long long ga = (unsigned long long)(uintptr_t)gsrc;
  uint32x4 g0;
  g0.x = 0x1u;                                        // count=1, load
  g0.y = lds_byte_off;                                // lds_addr
  g0.z = (unsigned)(ga & 0xffffffffu);                // global_addr[31:0]
  g0.w = (unsigned)((ga >> 32) & 0x1ffffffu) | (2u << 30);  // addr[56:32]|type=2
  int32x8 g1;
  g1[0] = (int)(3u << 16);                            // data_size = 8 bytes
  g1[1] = (int)((TDM_ELEMS & 0xffffu) << 16);         // tensor_dim0 lo16
  g1[2] = (int)((TDM_ELEMS >> 16) | (1u << 16));      // dim0 hi16 | tensor_dim1=1
  g1[3] = (int)((TDM_ELEMS & 0xffffu) << 16);         // tile_dim0 = 4096
  g1[4] = 0;                                          // tile_dim1/2 unused
  g1[5] = (int)TDM_ELEMS;                             // tensor_dim0_stride lo32
  g1[6] = 0;
  g1[7] = 0;
  int32x4 z4 = {0, 0, 0, 0};
#if defined(__clang_major__) && (__clang_major__ >= 23)
  int32x8 z8 = {0, 0, 0, 0, 0, 0, 0, 0};
  __builtin_amdgcn_tensor_load_to_lds(g0, g1, z4, z4, z8, 0);
#else
  __builtin_amdgcn_tensor_load_to_lds(g0, g1, z4, z4, 0);
#endif
#endif
}

__device__ __forceinline__ v16h load_frag(const _Float16* fb, int lane) {
  v8h lo = *(const v8h*)(fb + lane * 8);
  v8h hh = *(const v8h*)(fb + 256 + lane * 8);
  return __builtin_shufflevector(lo, hh, 0, 1, 2, 3, 4, 5, 6, 7,
                                 8, 9, 10, 11, 12, 13, 14, 15);
}

// ---------------------------------------------------------------------------
// Process one staged 4-tile chunk; C = weight-matrix index (cu >> 4), which
// is constant per chunk. kt is the OUTER (unroll-1) loop; per step we re-read
// the A fragment from LDS and feed FOUR independent WMMA chains (one per
// cu-tile). No runtime-indexed register arrays, bounded load lookahead.
// ---------------------------------------------------------------------------
template <int C>
__device__ __forceinline__ void process_chunk(
    const _Float16* __restrict__ Bl, const _Float16* __restrict__ aptr,
    const float* __restrict__ coefS, int lane, int wv, int hi, int ubase,
    float (&acc0)[8], float (&acc2)[8],
    float (&acc3a)[8], float (&acc3b)[8], float (&acc3c)[8]) {
  const int ebase = wv * 16 + hi * 8;                  // 8 consecutive edges
  v8f D[4];
#pragma unroll
  for (int tt = 0; tt < CHUNK_TILES; ++tt)
    D[tt] = (v8f){0.f, 0.f, 0.f, 0.f, 0.f, 0.f, 0.f, 0.f};
#pragma unroll 1
  for (int kt = 0; kt < 8; ++kt) {
    const _Float16* hp = aptr + kt * 32;               // A frag from LDS
    v8h alo = *(const v8h*)(hp);
    v8h ahi = *(const v8h*)(hp + 8);
    v16h Af = __builtin_shufflevector(alo, ahi, 0, 1, 2, 3, 4, 5, 6, 7,
                                      8, 9, 10, 11, 12, 13, 14, 15);
    const _Float16* fk = Bl + kt * FRAG_HALFS;
    v16h B0 = load_frag(fk + 0 * 8 * FRAG_HALFS, lane);
    v16h B1 = load_frag(fk + 1 * 8 * FRAG_HALFS, lane);
    v16h B2 = load_frag(fk + 2 * 8 * FRAG_HALFS, lane);
    v16h B3 = load_frag(fk + 3 * 8 * FRAG_HALFS, lane);
    D[0] = __builtin_amdgcn_wmma_f32_16x16x32_f16(false, Af, false, B0,
                                                  (short)0, D[0], false, false);
    D[1] = __builtin_amdgcn_wmma_f32_16x16x32_f16(false, Af, false, B1,
                                                  (short)0, D[1], false, false);
    D[2] = __builtin_amdgcn_wmma_f32_16x16x32_f16(false, Af, false, B2,
                                                  (short)0, D[2], false, false);
    D[3] = __builtin_amdgcn_wmma_f32_16x16x32_f16(false, Af, false, B3,
                                                  (short)0, D[3], false, false);
  }
#pragma unroll
  for (int tt = 0; tt < CHUNK_TILES; ++tt) {           // fold finished tiles
    const int u = ubase + tt;
    if (C == 3) {
      const float* p3 = &coefS[(3 * 16 + u) * TM + ebase];
      const float* p4 = &coefS[(4 * 16 + u) * TM + ebase];
      const float* p5 = &coefS[(5 * 16 + u) * TM + ebase];
      v4f a0 = *(const v4f*)p3, a1 = *(const v4f*)(p3 + 4);
      v4f b0 = *(const v4f*)p4, b1 = *(const v4f*)(p4 + 4);
      v4f c0 = *(const v4f*)p5, c1 = *(const v4f*)(p5 + 4);
#pragma unroll
      for (int r = 0; r < 8; ++r) {
        float d = D[tt][r];
        acc3a[r] += (r < 4 ? a0[r] : a1[r - 4]) * d;
        acc3b[r] += (r < 4 ? b0[r] : b1[r - 4]) * d;
        acc3c[r] += (r < 4 ? c0[r] : c1[r - 4]) * d;
      }
    } else {
      const float* cp = &coefS[(C * 16 + u) * TM + ebase];
      v4f cA = *(const v4f*)cp, cB = *(const v4f*)(cp + 4);
      float (&dst)[8] = (C == 2) ? acc2 : acc0;
#pragma unroll
      for (int r = 0; r < 8; ++r)
        dst[r] += (r < 4 ? cA[r] : cB[r - 4]) * D[tt][r];
    }
  }
}

// ---------------------------------------------------------------------------
// Kernel 2: fused FC-GEMM (WMMA f16->f32, B staged via TDM double buffer)
// + equivariant tensor product + atomic segment-sum scatter.
// 16 waves x 1 M-tile (16 edges) = 256 edges per workgroup.
// ---------------------------------------------------------------------------
__global__ __launch_bounds__(THREADS)
__attribute__((amdgpu_waves_per_eu(4)))
void conv_kernel(const int*   __restrict__ edge_src,
                 const int*   __restrict__ edge_dst,
                 const float* __restrict__ node_features,
                 const float* __restrict__ edge_sh,
                 const float* __restrict__ ell,
                 const float* __restrict__ fc_w1,
                 const _Float16* __restrict__ bswz,
                 const int*   __restrict__ nn,
                 float* __restrict__ out) {
  extern __shared__ __align__(16) char smem[];
  _Float16* hS    = (_Float16*)(smem + H_OFF);
  float*    coefS = (float*)(smem + COEF_OFF);
  float*    shY   = (float*)(smem + SHY_OFF);
  int*      dstS  = (int*)(smem + DST_OFF);
  float*    w1s   = (float*)(smem + W1_OFF);
  _Float16* Bbuf  = (_Float16*)(smem + B_OFF);

  const int t  = threadIdx.x;
  const int e0 = blockIdx.x * TM;
  const int lane = t & 31;
  const int wv   = t >> 5;
  const int hi   = lane >> 4;

#if HAS_TDM
  if (wv == 0)   // start DMA of B chunk 0 immediately; overlaps staging below
    tdm_load_chunk(bswz, (unsigned)(uintptr_t)(void*)Bbuf);
#endif

  for (int i = t; i < 3 * FC_HIDDEN; i += THREADS) w1s[i] = fc_w1[i];
  __syncthreads();

  const float inv_sqrt3 = 0.57735026918962576f;

  // ---- h = relu(ell @ w1) * (1/sqrt3)*sqrt2*(1/16), fp16, 2 threads/edge
  {
    int e = t >> 1, part = t & 1;
    float l0 = ell[(e0 + e) * 3 + 0];
    float l1 = ell[(e0 + e) * 3 + 1];
    float l2 = ell[(e0 + e) * 3 + 2];
    const float hs = inv_sqrt3 * 1.4142135623730951f * (1.0f / 16.0f);
    for (int k = part * 128; k < part * 128 + 128; k += 8) {
      v8h pack;
#pragma unroll
      for (int j = 0; j < 8; ++j) {
        float d = l0 * w1s[k + j] + l1 * w1s[256 + k + j] + l2 * w1s[512 + k + j];
        pack[j] = (_Float16)(fmaxf(d, 0.0f) * hs);
      }
      *(v8h*)&hS[e * HSTRIDE + k] = pack;       // ds_store_b128
    }
  }

  // ---- gather + coefficient tables, 2 threads/edge (8 u-slots each)
  {
    int e = t >> 1, part = t & 1, ge = e0 + e;
    float scale = rsqrtf((float)nn[0]);
    const float pw0 = 0.1767766952966369f;   // sqrt(1/32)
    const float pw1 = 0.3061862178478972f;   // sqrt(3/32)
    float C0 = pw0 * scale;
    float C1 = pw0 * inv_sqrt3 * scale;
    float C2 = pw1 * inv_sqrt3 * scale;
    float y0  = edge_sh[ge * 4 + 0];
    float y1a = edge_sh[ge * 4 + 1];
    float y1b = edge_sh[ge * 4 + 2];
    float y1c = edge_sh[ge * 4 + 3];
    if (part == 0) {
      shY[e * 3 + 0] = y1a; shY[e * 3 + 1] = y1b; shY[e * 3 + 2] = y1c;
      dstS[e] = edge_dst[ge];
    }
    const float* x = node_features + (size_t)edge_src[ge] * 64;
#pragma unroll
    for (int uu = 0; uu < 8; ++uu) {
      int u = part * 8 + uu;
      float s0 = x[u];
      float va = x[16 + u * 3 + 0];
      float vb = x[16 + u * 3 + 1];
      float vc = x[16 + u * 3 + 2];
      coefS[(0 * 16 + u) * TM + e] = C0 * y0 * s0;
      coefS[(1 * 16 + u) * TM + e] = C1 * (va * y1a + vb * y1b + vc * y1c);
      coefS[(2 * 16 + u) * TM + e] = C2 * s0;
      coefS[(3 * 16 + u) * TM + e] = C2 * y0 * va;
      coefS[(4 * 16 + u) * TM + e] = C2 * y0 * vb;
      coefS[(5 * 16 + u) * TM + e] = C2 * y0 * vc;
    }
  }
  __syncthreads();

  // per-wave A fragment base in LDS (row = this wave's M-tile row for lane)
  const _Float16* aptr = &hS[(wv * 16 + (lane & 15)) * HSTRIDE + hi * 16];

  float acc0[8] = {}, acc2[8] = {};
  float acc3a[8] = {}, acc3b[8] = {}, acc3c[8] = {};

  for (int ch = 0; ch < NCHUNKS; ++ch) {
    _Float16* Bl = Bbuf + (size_t)(ch & 1) * CHUNK_HALFS;
#if HAS_TDM
    if (wv == 0) __builtin_amdgcn_s_wait_tensorcnt(0);  // chunk ch landed
    __syncthreads();                                    // visible to all waves
    if (wv == 0 && ch + 1 < NCHUNKS)                    // prefetch next chunk
      tdm_load_chunk(bswz + (size_t)(ch + 1) * CHUNK_HALFS,
                     (unsigned)(uintptr_t)(void*)(Bbuf +
                         (size_t)((ch + 1) & 1) * CHUNK_HALFS));
#else
    __syncthreads();                                    // prev reads done
    {  // cooperative copy fallback
      const v16h* src = (const v16h*)(bswz + (size_t)ch * CHUNK_HALFS);
      v16h*       dst = (v16h*)Bl;
      for (int i = t; i < CHUNK_HALFS / 16; i += THREADS) dst[i] = src[i];
    }
    __syncthreads();
#endif
    const int ubase = (ch & 3) * 4;
    switch (ch >> 2) {   // c is uniform per chunk -> specialized bodies
      case 0: process_chunk<0>(Bl, aptr, coefS, lane, wv, hi, ubase,
                               acc0, acc2, acc3a, acc3b, acc3c); break;
      case 1: process_chunk<1>(Bl, aptr, coefS, lane, wv, hi, ubase,
                               acc0, acc2, acc3a, acc3b, acc3c); break;
      case 2: process_chunk<2>(Bl, aptr, coefS, lane, wv, hi, ubase,
                               acc0, acc2, acc3a, acc3b, acc3c); break;
      default: process_chunk<3>(Bl, aptr, coefS, lane, wv, hi, ubase,
                                acc0, acc2, acc3a, acc3b, acc3c); break;
    }
  }

  // ---- segment-sum scatter (global f32 atomics)
  const int col = lane & 15;
#pragma unroll
  for (int r = 0; r < 8; ++r) {
    int e = wv * 16 + hi * 8 + r;
    float* o = out + (size_t)dstS[e] * 64;
    float t2 = acc2[r];
    atomicAdd(o + col, acc0[r]);
    atomicAdd(o + 16 + col * 3 + 0, shY[e * 3 + 0] * t2 + acc3a[r]);
    atomicAdd(o + 16 + col * 3 + 1, shY[e * 3 + 1] * t2 + acc3b[r]);
    atomicAdd(o + 16 + col * 3 + 2, shY[e * 3 + 2] * t2 + acc3c[r]);
  }
}

extern "C" void kernel_launch(void* const* d_in, const int* in_sizes, int n_in,
                              void* d_out, int out_size, void* d_ws, size_t ws_size,
                              hipStream_t stream) {
  (void)in_sizes; (void)n_in; (void)ws_size;
  const int*   edge_src      = (const int*)d_in[0];
  const int*   edge_dst      = (const int*)d_in[1];
  const float* node_features = (const float*)d_in[2];
  const float* edge_sh       = (const float*)d_in[3];
  const float* ell           = (const float*)d_in[4];
  const float* fc_w1         = (const float*)d_in[5];
  const float* fc_w2         = (const float*)d_in[6];
  const int*   nn            = (const int*)d_in[7];
  float*    out  = (float*)d_out;
  _Float16* bswz = (_Float16*)d_ws;   // 512 KB scratch: swizzled fc_w2

  hipMemsetAsync(d_out, 0, (size_t)out_size * sizeof(float), stream);
  swizzle_w2_kernel<<<64, 256, 0, stream>>>(fc_w2, bswz);
  conv_kernel<<<N_EDGES / TM, THREADS, SMEM_TOTAL, stream>>>(
      edge_src, edge_dst, node_features, edge_sh, ell, fc_w1, bswz, nn, out);
}
